// MultiHeadSelfAttention_41472204210839
// MI455X (gfx1250) — matrix operand order repaired and checked by reading, not compiled, not run on hardware
//
#include <hip/hip_runtime.h>

// ---------------------------------------------------------------------------
// MHSA for MI455X (gfx1250): bf16 WMMA GEMMs + fused LN epilogue + flash attn
// with GLOBAL_LOAD_ASYNC_TO_LDS double-buffered staging.
// B=2, S=2048, D=1024, H=16, hd=64
// ---------------------------------------------------------------------------

#define B_DIM  2
#define S_LEN  2048
#define D_DIM  1024
#define H_NUM  16
#define HD     64
#define BS     (B_DIM * S_LEN)        // 4096 rows
#define LN_EPS 1e-5f

typedef __bf16 bf16;
typedef __attribute__((ext_vector_type(4)))  __bf16 bf16x4;
typedef __attribute__((ext_vector_type(8)))  __bf16 bf16x8;
typedef __attribute__((ext_vector_type(16))) __bf16 v16bf;
typedef __attribute__((ext_vector_type(8)))  float  v8f;
typedef __attribute__((ext_vector_type(4)))  float  f32x4;

// ---- CDNA5 async global->LDS DMA (ASYNCcnt-tracked) -----------------------
// One 16B transfer per lane. VDST VGPR carries the LDS byte offset; for
// generic pointers to __shared__ the LDS offset is the low 32 address bits
// (LDS aperture: tag in [63:32], offset in [31:0]).
__device__ __forceinline__ void async_b128(void* lds, const void* gaddr) {
  unsigned loff = (unsigned)(uintptr_t)lds;
  asm volatile("global_load_async_to_lds_b128 %0, %1, off"
               :: "v"(loff), "v"(gaddr) : "memory");
}

template <int N>
__device__ __forceinline__ void wait_async() {
  asm volatile("s_wait_asynccnt %0" :: "i"(N) : "memory");
}

// ---- WMMA helpers ---------------------------------------------------------

__device__ __forceinline__ v16bf concat8(bf16x8 lo, bf16x8 hi) {
  return __builtin_shufflevector(lo, hi, 0,1,2,3,4,5,6,7,8,9,10,11,12,13,14,15);
}

// A-matrix fragment, 16x32 bf16 (MxK). Lane l: row m=l&15, K = kb..kb+7 and
// kb+16..kb+23 with kb = 8*(l>>4). Source row-major [M][ld].
__device__ __forceinline__ v16bf load_a_frag(const bf16* base, int ld, int lane) {
  const int m  = lane & 15;
  const int kb = (lane >> 4) * 8;
  const bf16* p = base + m * ld + kb;
  return concat8(*(const bf16x8*)p, *(const bf16x8*)(p + 16));
}

// B-matrix fragment, 32x16 bf16 (KxN). Lane l: column n=l&15, K = kb..kb+7
// and kb+16..kb+23 with kb = 8*(l>>4). Source stored [N][ld] so a B column is contiguous K.
__device__ __forceinline__ v16bf load_b_frag(const bf16* base, int ld, int lane) {
  const int n  = lane & 15;
  const int kb = (lane >> 4) * 8;
  const bf16* p = base + n * ld + kb;
  return concat8(*(const bf16x8*)p, *(const bf16x8*)(p + 16));
}

__device__ __forceinline__ v8f wmma_bf16(v16bf a, v16bf b, v8f c) {
  return __builtin_amdgcn_wmma_f32_16x16x32_bf16(false, a, false, b,
                                                 (short)0, c, false, false);
}

// Reductions across the 16-lane half that owns a C-matrix row
// (rows 0-7 -> lanes 0-15, rows 8-15 -> lanes 16-31; xor bits 0..3 only).
__device__ __forceinline__ float half_max(float v) {
  v = fmaxf(v, __shfl_xor(v, 8, 32));
  v = fmaxf(v, __shfl_xor(v, 4, 32));
  v = fmaxf(v, __shfl_xor(v, 2, 32));
  v = fmaxf(v, __shfl_xor(v, 1, 32));
  return v;
}
__device__ __forceinline__ float half_sum(float v) {
  v += __shfl_xor(v, 8, 32);
  v += __shfl_xor(v, 4, 32);
  v += __shfl_xor(v, 2, 32);
  v += __shfl_xor(v, 1, 32);
  return v;
}

// ---- fp32 -> bf16 conversion (4-wide) -------------------------------------

__global__ void __launch_bounds__(256)
cvt_bf16_kernel(const f32x4* __restrict__ src, bf16x4* __restrict__ dst, int n4) {
  int i = blockIdx.x * 256 + threadIdx.x;
  if (i < n4) {
    f32x4 x = src[i];
    bf16x4 y;
    #pragma unroll
    for (int j = 0; j < 4; j++) y[j] = (bf16)x[j];
    dst[i] = y;
  }
}

// ---- Projection GEMM with fused per-64-column LayerNorm epilogue ----------
// C[m][n] = sum_k X[m][k] * W[n][k] + bias[n]; optional LN over the block's
// 64 columns (exactly one head-dim group). Output bf16.
// Block tile 256x64, wave tile 32x64 (2 A-frags x 4 B-frags = 8 WMMA/step,
// every B fragment reused twice).
__global__ void __launch_bounds__(256)
gemm_ln_kernel(const bf16* __restrict__ X, const bf16* __restrict__ W,
               const float* __restrict__ bias,
               const float* __restrict__ ln_scale, const float* __restrict__ ln_bias,
               int do_ln, bf16* __restrict__ out)
{
  __shared__ __align__(16) bf16 Xt[2][256][32];   // 32 KB (double-buffered)
  __shared__ __align__(16) bf16 Wt[2][64][32];    //  8 KB

  const int tid  = threadIdx.x;
  const int lane = tid & 31;
  const int wave = tid >> 5;
  const int half = lane >> 4;
  const int ln   = lane & 15;
  const int n0   = blockIdx.x * 64;
  const int m0   = blockIdx.y * 256;

  // async-stage one K-tile: 4 X-chunks + 1 W-chunk of 16B per thread
  auto stage = [&](int buf, int k0) {
    #pragma unroll
    for (int i = 0; i < 4; i++) {
      int c = tid + i * 256;                       // 1024 chunks of X
      int row = c >> 2, col = (c & 3) * 8;
      async_b128(&Xt[buf][row][col],
                 &X[(size_t)(m0 + row) * D_DIM + k0 + col]);
    }
    int row = tid >> 2, col = (tid & 3) * 8;       // 256 chunks of W
    async_b128(&Wt[buf][row][col],
               &W[(size_t)(n0 + row) * D_DIM + k0 + col]);
  };

  v8f acc[2][4] = {};

  auto compute = [&](int p) {
    __syncthreads();
    v16bf a0 = load_a_frag(&Xt[p][wave * 32][0],      32, lane);
    v16bf a1 = load_a_frag(&Xt[p][wave * 32 + 16][0], 32, lane);
    #pragma unroll
    for (int nt = 0; nt < 4; nt++) {
      v16bf bfrag = load_b_frag(&Wt[p][nt * 16][0], 32, lane);
      acc[0][nt] = wmma_bf16(a0, bfrag, acc[0][nt]);
      acc[1][nt] = wmma_bf16(a1, bfrag, acc[1][nt]);
    }
    __syncthreads();                    // all waves done with buf p
  };

  // steady state: stage tile k+1 while tile k is in flight, peel the last step
  stage(0, 0);
  int p = 0;
  for (int k0 = 0; k0 < D_DIM - 32; k0 += 32) {
    stage(p ^ 1, k0 + 32);              // 5 new async ops outstanding
    wait_async<5>();                    // tile k0 resident (in-order)
    if (k0 + 64 < D_DIM)                // deep prefetch past the double buffer
      __builtin_prefetch(&X[(size_t)(m0 + tid) * D_DIM + k0 + 64], 0, 3);
    compute(p);
    p ^= 1;
  }
  wait_async<0>();
  compute(p);

  // epilogue: bias (+ LayerNorm over the 64 columns of this block) -> bf16
  float bv[4], sc[4], sb[4];
  #pragma unroll
  for (int nt = 0; nt < 4; nt++) {
    int e = nt * 16 + ln;
    bv[nt] = bias[n0 + e];
    sc[nt] = do_ln ? ln_scale[e] : 1.0f;
    sb[nt] = do_ln ? ln_bias[e]  : 0.0f;
  }
  #pragma unroll
  for (int sub = 0; sub < 2; sub++) {
    #pragma unroll
    for (int r = 0; r < 8; r++) {
      float x[4];
      #pragma unroll
      for (int nt = 0; nt < 4; nt++) x[nt] = acc[sub][nt][r] + bv[nt];
      if (do_ln) {
        float s = x[0] + x[1] + x[2] + x[3];
        float q = x[0]*x[0] + x[1]*x[1] + x[2]*x[2] + x[3]*x[3];
        s = half_sum(s);
        q = half_sum(q);
        float mu  = s * (1.0f / 64.0f);
        float var = q * (1.0f / 64.0f) - mu * mu;
        float inv = rsqrtf(var + LN_EPS);
        #pragma unroll
        for (int nt = 0; nt < 4; nt++)
          x[nt] = (x[nt] - mu) * inv * sc[nt] + sb[nt];
      }
      int m = m0 + wave * 32 + sub * 16 + r + half * 8;
      #pragma unroll
      for (int nt = 0; nt < 4; nt++)
        out[(size_t)m * D_DIM + n0 + nt * 16 + ln] = (bf16)x[nt];
    }
  }
}

// ---- Flash attention over the quirk-reshaped heads ------------------------
// qp/kp/vp are bf16 [B*H][S][hd]. One block per (128 q-rows, b*H+h).
__global__ void __launch_bounds__(256)
attn_kernel(const bf16* __restrict__ qp, const bf16* __restrict__ kp,
            const bf16* __restrict__ vp, float* __restrict__ out)
{
  __shared__ __align__(16) bf16 Kt[2][64][HD];    // 16 KB, row-major [kv][e]
  __shared__ __align__(16) bf16 Vr[2][64][HD];    // 16 KB, raw V [kv][e]
  __shared__ __align__(16) bf16 Vt[HD][64];       //  8 KB, transposed [e][kv]
  __shared__ __align__(16) bf16 Pt[8][16][64];    // 16 KB per-wave P scratch

  const int tid  = threadIdx.x;
  const int lane = tid & 31;
  const int wave = tid >> 5;
  const int half = lane >> 4;
  const int ln   = lane & 15;
  const int q0   = blockIdx.x * 128;
  const int bh   = blockIdx.y;
  const size_t base = (size_t)bh * S_LEN * HD;

  // async-stage one kv-tile of K and raw V: 4 x 16B per thread
  auto stage = [&](int buf, int kv0) {
    const bf16* kb = kp + base + (size_t)kv0 * HD;
    const bf16* vb = vp + base + (size_t)kv0 * HD;
    #pragma unroll
    for (int i = 0; i < 2; i++) {
      int c = tid + i * 256;                       // 512 chunks each
      int row = c >> 3, col = (c & 7) * 8;
      async_b128(&Kt[buf][row][col], &kb[row * HD + col]);
      async_b128(&Vr[buf][row][col], &vb[row * HD + col]);
    }
  };

  // Q fragments for this wave's 16 rows, K-dim = hd = 64 -> 2 fragments
  v16bf qf[2];
  {
    const bf16* qb = qp + base + (size_t)(q0 + wave * 16) * HD;
    #pragma unroll
    for (int f = 0; f < 2; f++) qf[f] = load_a_frag(qb + f * 32, HD, lane);
  }

  v8f o[4] = {};
  float mrow[8], lrow[8];
  #pragma unroll
  for (int r = 0; r < 8; r++) { mrow[r] = -1e30f; lrow[r] = 0.0f; }
  const float scale = 0.125f;  // 1/sqrt(64)

  auto body = [&](int p) {
    __syncthreads();                    // Kt[p], Vr[p] visible

    // LDS->LDS transpose: Vr[p][kv][e] -> Vt[e][kv]
    #pragma unroll
    for (int i = 0; i < 2; i++) {
      int c = tid + i * 256;
      int row = c >> 3, col = (c & 7) * 8;
      bf16x8 v8 = *(const bf16x8*)&Vr[p][row][col];
      #pragma unroll
      for (int j = 0; j < 8; j++) Vt[col + j][row] = v8[j];
    }
    __syncthreads();                    // Vt ready

    // S = (Q_ln) (K_ln)^T : 16 x 64 per wave, K-dim = hd
    v8f s[4] = {};
    #pragma unroll
    for (int f = 0; f < 2; f++) {
      #pragma unroll
      for (int nt = 0; nt < 4; nt++) {
        v16bf kfrag = load_b_frag(&Kt[p][nt * 16][f * 32], HD, lane);
        s[nt] = wmma_bf16(qf[f], kfrag, s[nt]);
      }
    }

    // online softmax per row (8 rows per lane-half)
    #pragma unroll
    for (int r = 0; r < 8; r++) {
      float pm = -1e30f;
      #pragma unroll
      for (int nt = 0; nt < 4; nt++) {
        s[nt][r] *= scale;
        pm = fmaxf(pm, s[nt][r]);
      }
      float rowmax = half_max(pm);
      float mnew   = fmaxf(mrow[r], rowmax);
      float corr   = __expf(mrow[r] - mnew);
      float psum   = 0.0f;
      #pragma unroll
      for (int nt = 0; nt < 4; nt++) {
        float pe = __expf(s[nt][r] - mnew);
        s[nt][r] = pe;
        psum += pe;
      }
      lrow[r] = lrow[r] * corr + half_sum(psum);
      mrow[r] = mnew;
      #pragma unroll
      for (int nt = 0; nt < 4; nt++) o[nt][r] *= corr;
    }

    // C-layout P -> bf16 A-layout via per-wave LDS scratch (no barrier needed)
    #pragma unroll
    for (int r = 0; r < 8; r++) {
      int m = r + half * 8;
      #pragma unroll
      for (int nt = 0; nt < 4; nt++)
        Pt[wave][m][nt * 16 + ln] = (bf16)s[nt][r];
    }
    v16bf pf[2];
    #pragma unroll
    for (int f = 0; f < 2; f++)
      pf[f] = load_a_frag(&Pt[wave][0][f * 32], 64, lane);

    // O += P @ V, K-dim = kv tile (64)
    #pragma unroll
    for (int f = 0; f < 2; f++) {
      #pragma unroll
      for (int nt = 0; nt < 4; nt++) {
        v16bf vfrag = load_b_frag(&Vt[nt * 16][f * 32], 64, lane);
        o[nt] = wmma_bf16(pf[f], vfrag, o[nt]);
      }
    }
    __syncthreads();                    // done reading Kt[p]/Vr[p]/Vt
  };

  // steady state + peeled last tile (straight-line waits, no branches)
  stage(0, 0);
  int p = 0;
  for (int kv0 = 0; kv0 < S_LEN - 64; kv0 += 64) {
    stage(p ^ 1, kv0 + 64);             // 4 new async ops outstanding
    wait_async<4>();                    // current tile resident
    body(p);
    p ^= 1;
  }
  wait_async<0>();
  body(p);

  // normalize and scatter with transpose(0,2,1,3) fold: out[b][s][h*hd+e]
  const int b = bh / H_NUM, h = bh % H_NUM;
  #pragma unroll
  for (int r = 0; r < 8; r++) {
    int m = q0 + wave * 16 + r + half * 8;
    float invl = 1.0f / lrow[r];
    #pragma unroll
    for (int nt = 0; nt < 4; nt++) {
      int e = nt * 16 + ln;
      out[((size_t)b * S_LEN + m) * D_DIM + h * HD + e] = o[nt][r] * invl;
    }
  }
}

// ---------------------------------------------------------------------------

extern "C" void kernel_launch(void* const* d_in, const int* in_sizes, int n_in,
                              void* d_out, int out_size, void* d_ws, size_t ws_size,
                              hipStream_t stream) {
  (void)in_sizes; (void)n_in; (void)out_size; (void)ws_size;
  const float* q   = (const float*)d_in[0];
  const float* k   = (const float*)d_in[1];
  const float* v   = (const float*)d_in[2];
  const float* Wq  = (const float*)d_in[3];
  const float* bq  = (const float*)d_in[4];
  const float* Wk  = (const float*)d_in[5];
  const float* bk  = (const float*)d_in[6];
  const float* Wv  = (const float*)d_in[7];
  const float* bv  = (const float*)d_in[8];
  const float* qsc = (const float*)d_in[9];
  const float* qbi = (const float*)d_in[10];
  const float* ksc = (const float*)d_in[11];
  const float* kbi = (const float*)d_in[12];
  float* out = (float*)d_out;

  const int NX = BS * D_DIM;       // 4,194,304 activation elems per tensor
  const int NW = D_DIM * D_DIM;    // 1,048,576 weight elems per tensor

  char* ws = (char*)d_ws;
  bf16* Xq = (bf16*)(ws);
  bf16* Xk = Xq + NX;
  bf16* Xv = Xk + NX;
  bf16* Gq = Xv + NX;              // bf16 weights
  bf16* Gk = Gq + NW;
  bf16* Gv = Gk + NW;
  bf16* QP = Gv + NW;              // LN'd projections, bf16 [B*H][S][hd]
  bf16* KP = QP + NX;
  bf16* VP = KP + NX;

  dim3 blk(256);
  dim3 cvtA((NX / 4 + 255) / 256), cvtW((NW / 4 + 255) / 256);
  cvt_bf16_kernel<<<cvtA, blk, 0, stream>>>((const f32x4*)q,  (bf16x4*)Xq, NX / 4);
  cvt_bf16_kernel<<<cvtA, blk, 0, stream>>>((const f32x4*)k,  (bf16x4*)Xk, NX / 4);
  cvt_bf16_kernel<<<cvtA, blk, 0, stream>>>((const f32x4*)v,  (bf16x4*)Xv, NX / 4);
  cvt_bf16_kernel<<<cvtW, blk, 0, stream>>>((const f32x4*)Wq, (bf16x4*)Gq, NW / 4);
  cvt_bf16_kernel<<<cvtW, blk, 0, stream>>>((const f32x4*)Wk, (bf16x4*)Gk, NW / 4);
  cvt_bf16_kernel<<<cvtW, blk, 0, stream>>>((const f32x4*)Wv, (bf16x4*)Gv, NW / 4);

  dim3 ggrid(D_DIM / 64, BS / 256);          // (16, 16)
  gemm_ln_kernel<<<ggrid, blk, 0, stream>>>(Xq, Gq, bq, qsc, qbi, 1, QP);
  gemm_ln_kernel<<<ggrid, blk, 0, stream>>>(Xk, Gk, bk, ksc, kbi, 1, KP);
  gemm_ln_kernel<<<ggrid, blk, 0, stream>>>(Xv, Gv, bv, qsc, qbi, 0, VP);

  dim3 agrid(S_LEN / 128, B_DIM * H_NUM);    // (16, 32)
  attn_kernel<<<agrid, blk, 0, stream>>>(QP, KP, VP, out);
}
